// Model_42657615184404
// MI455X (gfx1250) — matrix operand order repaired and checked by reading, not compiled, not run on hardware
//
#include <hip/hip_runtime.h>
#include <math.h>

// ---------------------------------------------------------------------------
// MI455X (gfx1250) implementation of the Gaunt-TP equivariant GNN forward.
// Core: fused per-edge grid tensor product using V_WMMA_F32_16X16X32_BF16,
// with the constant back-projection operand staged into LDS by the Tensor
// Data Mover (TENSOR_LOAD_TO_LDS, TENSORcnt) in iterate mode (row padding
// for conflict-free ds_load_b128 B-fragment reads).
// ---------------------------------------------------------------------------

typedef __attribute__((ext_vector_type(16))) __bf16 v16bf;
typedef __attribute__((ext_vector_type(8)))  float  v8f;
typedef __attribute__((ext_vector_type(4)))  unsigned int v4u;

typedef __attribute__((ext_vector_type(4))) unsigned int u32x4;
typedef __attribute__((ext_vector_type(4))) int          i32x4;
typedef __attribute__((ext_vector_type(8))) int          i32x8;

union Frag16 { v16bf v; v4u u[2]; };

#define PI_D 3.14159265358979323846

// ======================= TDM: global rows -> padded LDS ====================
// Builds a D# (CDNA5 ISA sect 8.3/8.4) for a 2D bf16 tile and
// issues TENSOR_LOAD_TO_LDS in iterate mode: tile = 1 row x rowElems,
// iterated `rows` times with lds_addr_increment = ldsRowElems (row padding).
__device__ __forceinline__ void tdm_load_rows_to_lds(
    const void* gsrc, unsigned ldsByteAddr,
    int rows, int rowElems, int ldsRowElems)
{
    unsigned long long ga = (unsigned long long)(uintptr_t)gsrc;

    u32x4 g0;
    g0[0] = 1u;                                        // count=1 valid user D#
    g0[1] = ldsByteAddr;                               // lds_addr (bytes)
    g0[2] = (unsigned)(ga & 0xFFFFFFFFu);              // global_addr[31:0]
    g0[3] = (unsigned)((ga >> 32) & 0x01FFFFFFu)       // global_addr[56:32]
          | (2u << 30);                                // type = 2 ("image")

    i32x8 g1;
    g1[0] = (1 << 16) | (1 << 19);                     // data_size=2B, iterate_enable
    g1[1] = (rowElems & 0xFFFF) << 16;                 // tensor_dim0[15:0]
    g1[2] = (rows & 0xFFFF) << 16;                     // dim0 hi=0 | tensor_dim1[15:0]
    g1[3] = (rowElems & 0xFFFF) << 16;                 // dim1 hi=0 | tile_dim0
    g1[4] = 1;                                         // tile_dim1 = 1 row / iter
    g1[5] = rowElems;                                  // tensor_dim0_stride[31:0]
    g1[6] = (rowElems & 0xFFFF) << 16;                 // stride hi=0 | dim1_stride lo
    g1[7] = 0;

    i32x4 g2;
    g2[0] = 0;                                         // tensor_dim2 (unused)
    g2[1] = ldsRowElems;                               // lds_addr_increment / iter
    g2[2] = rowElems;                                  // global_addr_increment / iter
    g2[3] = (rows - 1) << 16;                          // iterate_count

    i32x4 g3 = {0, 0, 0, 0};

#if __has_include(<hip/amd_detail/amd_gfx1250_TDM.h>)
    // amdgpu-toolchain (clang-23 / therock): 6-arg builtin
    i32x8 g4 = {0, 0, 0, 0, 0, 0, 0, 0};
    __builtin_amdgcn_tensor_load_to_lds(g0, g1, g2, g3, g4, 0);
#else
    // ROCm 7.2 (clang-22): 5-arg builtin
    __builtin_amdgcn_tensor_load_to_lds(g0, g1, g2, g3, 0);
#endif
}

// =============================== grid init =================================
// Gauss-Legendre (n=20) x uniform alpha (19); real spherical harmonics to
// l=5 packed directly into the WMMA operand layouts:
//   YGB  [384][32] : row = grid col, col = K (16 used)      -> phase-1 B frags
//   YGzB [48][384] : row = output sh row (36 used), col = K -> phase-2 B frags
__global__ void grid_init_kernel(__bf16* __restrict__ YGB,
                                 __bf16* __restrict__ YGzB,
                                 float*  __restrict__ gridW)
{
    int g = threadIdx.x;              // 0..383 (380 valid)
    bool valid = g < 380;
    int b = g / 19, a = g % 19;

    double Y[36];
    for (int i = 0; i < 36; ++i) Y[i] = 0.0;
    double wq = 0.0;

    if (valid) {
        const int n = 20;
        double x = cos(PI_D * (b + 0.75) / (n + 0.5));
        for (int it = 0; it < 40; ++it) {          // Newton on P_20
            double p0 = 1.0, p1 = x;
            for (int k = 2; k <= n; ++k) {
                double p2 = ((2.0 * k - 1.0) * x * p1 - (k - 1.0) * p0) / k;
                p0 = p1; p1 = p2;
            }
            double dp = n * (x * p1 - p0) / (x * x - 1.0);
            x -= p1 / dp;
        }
        {   // quadrature weight
            double p0 = 1.0, p1 = x;
            for (int k = 2; k <= n; ++k) {
                double p2 = ((2.0 * k - 1.0) * x * p1 - (k - 1.0) * p0) / k;
                p0 = p1; p1 = p2;
            }
            double dp = n * (x * p1 - p0) / (x * x - 1.0);
            wq = 2.0 / ((1.0 - x * x) * dp * dp);
        }
        double P[6][6];
        double s = sqrt(fmax(0.0, 1.0 - x * x));
        P[0][0] = 1.0;
        for (int m = 1; m <= 5; ++m) P[m][m] = -(2.0 * m - 1.0) * s * P[m - 1][m - 1];
        for (int m = 0; m < 5; ++m)  P[m + 1][m] = (2.0 * m + 1.0) * x * P[m][m];
        for (int m = 0; m <= 5; ++m)
            for (int l = m + 2; l <= 5; ++l)
                P[l][m] = ((2.0 * l - 1.0) * x * P[l - 1][m] - (l - 1.0 + m) * P[l - 2][m]) / (l - m);

        double alpha = 2.0 * PI_D * a / 19.0;
        double fact[11]; fact[0] = 1.0;
        for (int i = 1; i <= 10; ++i) fact[i] = fact[i - 1] * i;

        int r = 0;
        for (int l = 0; l <= 5; ++l)
            for (int m = -l; m <= l; ++m, ++r) {
                int am = m < 0 ? -m : m;
                double K = sqrt((2.0 * l + 1.0) / (4.0 * PI_D) * fact[l - am] / fact[l + am]);
                double ang = (m > 0) ? sqrt(2.0) * cos((double)m * alpha)
                           : (m < 0) ? sqrt(2.0) * sin((double)am * alpha)
                                     : 1.0;
                Y[r] = K * P[l][am] * ang;
            }
    }

    for (int k = 0; k < 32; ++k)
        YGB[(size_t)g * 32 + k] = (__bf16)(float)((k < 16) ? Y[k] : 0.0);
    for (int nr = 0; nr < 48; ++nr)
        YGzB[(size_t)nr * 384 + g] = (__bf16)(float)((nr < 36) ? Y[nr] : 0.0);
    gridW[g] = (float)(wq * (2.0 * PI_D / 19.0));
}

// ============================ edge spherical harmonics =====================
__global__ void edge_sh_kernel(const float* __restrict__ pos,
                               const int* __restrict__ snd,
                               const int* __restrict__ rcv,
                               float* __restrict__ sh, int E)
{
    int e = blockIdx.x * blockDim.x + threadIdx.x;
    if (e >= E) return;
    int s = snd[e], r = rcv[e];
    float rx = pos[r * 3 + 0] - pos[s * 3 + 0];
    float ry = pos[r * 3 + 1] - pos[s * 3 + 1];
    float rz = pos[r * 3 + 2] - pos[s * 3 + 2];
    float inv = rsqrtf(rx * rx + ry * ry + rz * rz + 1e-9f);
    float x = rx * inv, y = ry * inv, z = rz * inv;
    float* o = sh + (size_t)e * 16;
    o[0]  = 0.f;
    o[1]  = 0.48860251f * y;  o[2] = 0.48860251f * z;  o[3] = 0.48860251f * x;
    o[4]  = 1.09254843f * x * y;
    o[5]  = 1.09254843f * y * z;
    o[6]  = 0.31539157f * (3.f * z * z - 1.f);
    o[7]  = 1.09254843f * x * z;
    o[8]  = 0.54627421f * (x * x - y * y);
    o[9]  = 0.59004359f * y * (3.f * x * x - y * y);
    o[10] = 2.89061144f * x * y * z;
    o[11] = 0.45704579f * y * (5.f * z * z - 1.f);
    o[12] = 0.37317633f * z * (5.f * z * z - 3.f);
    o[13] = 0.45704579f * x * (5.f * z * z - 1.f);
    o[14] = 1.44530572f * z * (x * x - y * y);
    o[15] = 0.59004359f * x * (x * x - 3.f * y * y);
}

// ====================== coefficient packing (bf16, K padded to 32) =========
__global__ void ycoef_kernel(const float* __restrict__ sh,
                             const float* w1, const float* w2, const float* w3,
                             __bf16* __restrict__ out, int E)
{
    int e = blockIdx.x * blockDim.x + threadIdx.x;
    if (e >= E) return;
    const float* s = sh + (size_t)e * 16;
    __bf16* o = out + (size_t)e * 32;
    float a1 = w1 ? w1[0] : 0.f, a2 = w2 ? w2[0] : 0.f, a3 = w3 ? w3[0] : 0.f;
    o[0] = (__bf16)0.f;
    for (int m = 0; m < 3; ++m) o[1 + m] = (__bf16)(a1 * s[1 + m]);
    for (int m = 0; m < 5; ++m) o[4 + m] = (__bf16)(a2 * s[4 + m]);
    for (int m = 0; m < 7; ++m) o[9 + m] = (__bf16)(a3 * s[9 + m]);
    for (int k = 16; k < 32; ++k) o[k] = (__bf16)0.f;
}

__global__ void xcoef_kernel(const float* __restrict__ feat,
                             const int* __restrict__ snd,
                             const float* w0, const float* w1, const float* w2,
                             int c0, int c1, int c2,
                             int o0, int o1, int o2, int prevDim,
                             __bf16* __restrict__ out, int E)
{
    int e = blockIdx.x * blockDim.x + threadIdx.x;
    if (e >= E) return;
    const float* f = feat + (size_t)snd[e] * prevDim;
    __bf16* o = out + (size_t)e * 32;
    float v[9];
    for (int i = 0; i < 9; ++i) v[i] = 0.f;
    if (w0) { float a = 0.f; for (int c = 0; c < c0; ++c) a += w0[c] * f[o0 + c]; v[0] = a; }
    if (w1) for (int m = 0; m < 3; ++m) {
        float a = 0.f; for (int c = 0; c < c1; ++c) a += w1[c] * f[o1 + c * 3 + m]; v[1 + m] = a;
    }
    if (w2) for (int m = 0; m < 5; ++m) {
        float a = 0.f; for (int c = 0; c < c2; ++c) a += w2[c] * f[o2 + c * 5 + m]; v[4 + m] = a;
    }
    for (int k = 0; k < 9;  ++k) o[k] = (__bf16)v[k];
    for (int k = 9; k < 32; ++k) o[k] = (__bf16)0.f;
}

// ============================ fused Gaunt grid TP ==========================
// 2 waves/block, one 16-edge tile per wave. At launch each wave TDM-stages
// half of YGzB (48x384 bf16) into LDS with a 16B row pad (conflict-free
// b128 fragment reads). Phase 1: fx,fy via 2x24 WMMAs sharing B frags,
// g = fx*fy*W packed bf16 into gLds (overlapping the TDM). Phase 2 after
// s_wait_tensorcnt + barrier: z = g @ YGz^T via 12x3 WMMAs, A from gLds,
// B from the TDM-staged tile.
__global__ __launch_bounds__(64) void gaunt_wmma_kernel(
    const __bf16* __restrict__ xcoefB,   // [E][32]
    const __bf16* __restrict__ ycoefB,   // [E][32]
    const __bf16* __restrict__ YGB,      // [384][32]
    const __bf16* __restrict__ YGzB,     // [48][384]
    const float*  __restrict__ gridW,    // [384]
    float* __restrict__ zOut,            // [E][36]
    int E)
{
    __shared__ __align__(16) __bf16 gLds[2][16][392];
    __shared__ __align__(16) __bf16 ygzLds[48 * 392];

    const int wave = threadIdx.x >> 5;   // 0..1
    const int lane = threadIdx.x & 31;
    const int hh   = lane >> 4;          // half-wave select
    const int lm   = lane & 15;

    int tile = blockIdx.x * 2 + wave;
    long e0 = (long)tile * 16;
    const bool active = (e0 < E);

    // ---- kick off TDM: this wave stages rows [wave*24, wave*24+24) --------
    {
        unsigned ldsBase = (unsigned)(uintptr_t)(void*)&ygzLds[0];
        tdm_load_rows_to_lds(YGzB + (size_t)wave * 24 * 384,
                             ldsBase + (unsigned)wave * 24 * 392 * 2,
                             /*rows=*/24, /*rowElems=*/384, /*ldsRowElems=*/392);
    }

    const v8f vzero = {};

    if (active) {
        long erow = e0 + lm; if (erow > E - 1) erow = E - 1;

        // A fragments (interleaved K pattern: lanes<16 hold K{0..7,16..23})
        Frag16 aX, aY;
        {
            const v4u* px = (const v4u*)(xcoefB + (size_t)erow * 32);
            const v4u* py = (const v4u*)(ycoefB + (size_t)erow * 32);
            aX.u[0] = px[hh];     aX.u[1] = px[2 + hh];
            aY.u[0] = py[hh];     aY.u[1] = py[2 + hh];
        }
        __builtin_prefetch(xcoefB + (size_t)(e0 + 32) * 32, 0, 1);
        __builtin_prefetch(ycoefB + (size_t)(e0 + 32) * 32, 0, 1);

        // ---- phase 1: grid values -> gLds --------------------------------
        for (int gt = 0; gt < 24; ++gt) {
            int col = gt * 16 + lm;
            Frag16 bg;   // interleaved K pattern: lane half h holds K{8h..8h+7,16+8h..16+8h+7}
            {
                const v4u* pb = (const v4u*)(YGB + (size_t)col * 32);
                bg.u[0] = pb[hh]; bg.u[1] = pb[2 + hh];
            }
            v8f fx = __builtin_amdgcn_wmma_f32_16x16x32_bf16(
                false, aX.v, false, bg.v, (short)0, vzero, false, false);
            v8f fy = __builtin_amdgcn_wmma_f32_16x16x32_bf16(
                false, aY.v, false, bg.v, (short)0, vzero, false, false);
            float w = gridW[col];
#pragma unroll
            for (int r = 0; r < 8; ++r) {
                float gg = fx[r] * fy[r] * w;           // row M = r + 8*hh
                gLds[wave][r + hh * 8][col] = (__bf16)gg;
            }
        }
    }

    // ---- TDM completion + make staged tile visible to both waves ---------
    __builtin_amdgcn_s_wait_tensorcnt(0);
    __syncthreads();

    if (active) {
        // ---- phase 2: z = g @ YGz^T (B from TDM-staged LDS) --------------
        v8f acc0 = {}, acc1 = {}, acc2 = {};
        for (int kt = 0; kt < 12; ++kt) {
            Frag16 ag;   // A from gLds, blocked pattern, row = edge (lm)
            {
                const v4u* pg = (const v4u*)&gLds[wave][lm][kt * 32];
                ag.u[0] = pg[2 * hh]; ag.u[1] = pg[2 * hh + 1];
            }
#pragma unroll
            for (int nt = 0; nt < 3; ++nt) {
                Frag16 bz;   // B fragment, blocked pattern, row = out col
                const v4u* pz = (const v4u*)&ygzLds[(size_t)(nt * 16 + lm) * 392
                                                    + kt * 32 + hh * 16];
                bz.u[0] = pz[0]; bz.u[1] = pz[1];
                if (nt == 0)
                    acc0 = __builtin_amdgcn_wmma_f32_16x16x32_bf16(
                        false, ag.v, false, bz.v, (short)0, acc0, false, false);
                else if (nt == 1)
                    acc1 = __builtin_amdgcn_wmma_f32_16x16x32_bf16(
                        false, ag.v, false, bz.v, (short)0, acc1, false, false);
                else
                    acc2 = __builtin_amdgcn_wmma_f32_16x16x32_bf16(
                        false, ag.v, false, bz.v, (short)0, acc2, false, false);
            }
        }

        // ---- store z (36 valid of 48 cols) -------------------------------
#pragma unroll
        for (int nt = 0; nt < 3; ++nt) {
            int col = nt * 16 + lm;
            if (col < 36) {
                v8f a = (nt == 0) ? acc0 : (nt == 1) ? acc1 : acc2;
#pragma unroll
                for (int r = 0; r < 8; ++r) {
                    long e = e0 + r + hh * 8;
                    if (e < E) zOut[(size_t)e * 36 + col] = a[r];
                }
            }
        }
    }
}

// ============================ aggregation ==================================
__global__ void scatter_pass_kernel(const float* __restrict__ feat,
                                    const int* __restrict__ snd,
                                    const int* __restrict__ rcv,
                                    float* __restrict__ aggPass,
                                    int E, int prevDim)
{
    long gid = (long)blockIdx.x * blockDim.x + threadIdx.x;
    long e = gid >> 5; int c0 = (int)(gid & 31);
    if (e >= E) return;
    int s = snd[e], r = rcv[e];
    for (int c = c0; c < prevDim; c += 32)
        atomicAdd(&aggPass[(size_t)r * prevDim + c], feat[(size_t)s * prevDim + c]);
}

__global__ void scatter_tp_kernel(const float* __restrict__ zBuf,
                                  const int* __restrict__ rcv,
                                  float* __restrict__ aggTP,
                                  int E, int t)
{
    long gid = (long)blockIdx.x * blockDim.x + threadIdx.x;
    long e = gid >> 5; int j0 = (int)(gid & 31);
    if (e >= E) return;
    int r = rcv[e];
    for (int j = j0; j < 36; j += 32)
        atomicAdd(&aggTP[(size_t)r * 108 + t * 36 + j], zBuf[(size_t)e * 36 + j]);
}

// ============================ node update ==================================
struct KeyD   { int l, parity, mul, off, prevMul, prevOff, tpm; };
struct LayerD { int nkeys, prevDim, outDim; KeyD k[6]; };
struct NodeW  { const float* pre[6]; const float* post[6]; const float* shw[6]; };

__global__ void node_init_kernel(float* feat, int N)
{
    int n = blockIdx.x * blockDim.x + threadIdx.x;
    if (n < N) feat[n] = 1.0f;
}

__global__ void node_update_kernel(const float* __restrict__ aggPass,
                                   const float* __restrict__ aggTP,
                                   const float* __restrict__ prevFeat,
                                   float* __restrict__ outFeat,
                                   LayerD D, NodeW W, int N)
{
    int n = blockIdx.x * blockDim.x + threadIdx.x;
    if (n >= N) return;
    const float inv_denom = 1.0f / 1.5f;
    const int lo[6] = {0, 1, 4, 9, 16, 25};

    for (int kk = 0; kk < D.nkeys; ++kk) {
        KeyD K = D.k[kk];
        int dim = 2 * K.l + 1;
        int tpidx[3], ntp = 0;
        for (int t = 0; t < 3; ++t) if ((K.tpm >> t) & 1) tpidx[ntp++] = t;

        float h[48];
        const float* pw = W.pre[kk];
        for (int o = 0; o < K.mul; ++o)
            for (int m = 0; m < dim; ++m) {
                float acc = 0.f;
                if (pw) {
                    for (int i = 0; i < K.prevMul; ++i)
                        acc += pw[i * K.mul + o] *
                               aggPass[(size_t)n * D.prevDim + K.prevOff + i * dim + m];
                    int wi = K.prevMul;
                    for (int j = 0; j < ntp; ++j, ++wi)
                        acc += pw[wi * K.mul + o] *
                               aggTP[(size_t)n * 108 + tpidx[j] * 36 + lo[K.l] + m];
                }
                h[o * dim + m] = acc * inv_denom;
            }
        if (K.l == 0) {
            for (int o = 0; o < K.mul; ++o) {
                float xv = h[o];
                h[o] = (K.parity > 0)
                     ? 0.5f * xv * (1.f + tanhf(0.79788456f * (xv + 0.044715f * xv * xv * xv)))
                     : tanhf(xv);
            }
        }
        const float* qw = W.post[kk];
        const float* sw = W.shw[kk];
        for (int o = 0; o < K.mul; ++o)
            for (int m = 0; m < dim; ++m) {
                float acc = 0.f;
                if (qw) for (int i = 0; i < K.mul; ++i)
                    acc += qw[i * K.mul + o] * h[i * dim + m];
                float sc = 0.f;
                if (sw) for (int i = 0; i < K.prevMul; ++i)
                    sc += sw[i * K.mul + o] *
                          prevFeat[(size_t)n * D.prevDim + K.prevOff + i * dim + m];
                outFeat[(size_t)n * D.outDim + K.off + o * dim + m] = acc + sc;
            }
    }
}

// ============================ readout ======================================
__global__ void readout_kernel(const float* __restrict__ feat,
                               float* __restrict__ out, int N, int G)
{
    __shared__ float pred[64];
    int t = threadIdx.x;                 // 64 threads: (graph, col)
    int g = t >> 3, c = t & 7;
    int per = N / G;
    float s = 0.f;
    if (g < G)
        for (int n = g * per; n < (g + 1) * per; ++n)
            s += feat[(size_t)n * 8 + c];
    pred[t] = s;
    __syncthreads();
    if (g < G) {
        if (c == 0) {
            float odd = pred[g * 8 + 0], e1 = pred[g * 8 + 1];
            out[g * 8 + 0] =  odd * e1;
            out[g * 8 + 1] = -odd * e1;
        } else if (c >= 2) {
            out[g * 8 + c] = pred[g * 8 + c];
        }
    }
}

// ============================ host orchestration ===========================
extern "C" void kernel_launch(void* const* d_in, const int* in_sizes, int n_in,
                              void* d_out, int out_size, void* d_ws, size_t ws_size,
                              hipStream_t stream)
{
    const float* positions = (const float*)d_in[0];
    const int*   senders   = (const int*)d_in[1];
    const int*   receivers = (const int*)d_in[2];
    int N = in_sizes[0] / 3;
    int E = in_sizes[1];
    int G = in_sizes[3];

    // param leaves (jax sorted-dict flatten order) start at d_in[4]
    auto leaf = [&](int i) -> const float* {
        return (i >= 0 && i < n_in) ? (const float*)d_in[i] : (const float*)nullptr;
    };

    // ---- workspace carve --------------------------------------------------
    char* p = (char*)d_ws;
    auto alloc = [&](size_t bytes) -> char* {
        char* q = p; p += (bytes + 255) & ~(size_t)255; return q;
    };
    __bf16* YGB    = (__bf16*)alloc(384 * 32 * 2);
    __bf16* YGzB   = (__bf16*)alloc(48 * 384 * 2);
    float*  gridW  = (float*)alloc(384 * 4);
    float*  shBuf  = (float*)alloc((size_t)E * 16 * 4);
    __bf16* xcoefB = (__bf16*)alloc((size_t)E * 32 * 2);
    __bf16* ycoefB = (__bf16*)alloc((size_t)E * 32 * 2);
    float*  zBuf   = (float*)alloc((size_t)E * 36 * 4);
    float*  nodeA  = (float*)alloc((size_t)N * 192 * 4);
    float*  nodeB  = (float*)alloc((size_t)N * 192 * 4);
    float*  aggPass= (float*)alloc((size_t)N * 192 * 4);
    float*  aggTP  = (float*)alloc((size_t)N * 108 * 4);
    if ((size_t)(p - (char*)d_ws) > ws_size) return;

    grid_init_kernel<<<1, 384, 0, stream>>>(YGB, YGzB, gridW);
    edge_sh_kernel<<<(E + 255) / 256, 256, 0, stream>>>(positions, senders, receivers, shBuf, E);
    node_init_kernel<<<(N + 255) / 256, 256, 0, stream>>>(nodeA, N);

    // ---- per-layer structure tables --------------------------------------
    LayerD D[3];
    D[0] = {6, 1, 192, {
        {0, +1, 32,   0, 1, 0, 0b001},
        {0, -1, 32,  32, 0, 0, 0b110},
        {1, +1,  8,  64, 0, 0, 0b110},
        {1, -1,  8,  88, 0, 0, 0b001},
        {2, +1,  8, 112, 0, 0, 0b001},
        {2, -1,  8, 152, 0, 0, 0b110}}};
    D[1] = {6, 192, 192, {
        {0, +1, 32,   0, 32,   0, 0b001},
        {0, -1, 32,  32, 32,  32, 0b110},
        {1, +1,  8,  64,  8,  64, 0b110},
        {1, -1,  8,  88,  8,  88, 0b001},
        {2, +1,  8, 112,  8, 112, 0b001},
        {2, -1,  8, 152,  8, 152, 0b110}}};
    D[2] = {2, 192, 8, {
        {0, -1, 1, 0, 32, 32, 0b110},
        {0, +1, 7, 1, 32,  0, 0b001},
        {0, 0, 0, 0, 0, 0, 0}, {0, 0, 0, 0, 0, 0, 0},
        {0, 0, 0, 0, 0, 0, 0}, {0, 0, 0, 0, 0, 0, 0}}};

    NodeW Wl[3];
    Wl[0] = {{leaf(11), leaf(10), leaf(13), leaf(12), leaf(15), leaf(14)},
             {leaf(5),  leaf(4),  leaf(7),  leaf(6),  leaf(9),  leaf(8)},
             {leaf(16), nullptr, nullptr, nullptr, nullptr, nullptr}};
    Wl[1] = {{leaf(32), leaf(31), leaf(34), leaf(33), leaf(36), leaf(35)},
             {leaf(26), leaf(25), leaf(28), leaf(27), leaf(30), leaf(29)},
             {leaf(38), leaf(37), leaf(40), leaf(39), leaf(42), leaf(41)}};
    Wl[2] = {{leaf(60), leaf(61), nullptr, nullptr, nullptr, nullptr},
             {leaf(58), leaf(59), nullptr, nullptr, nullptr, nullptr},
             {leaf(62), leaf(63), nullptr, nullptr, nullptr, nullptr}};

    const float* XW[3][3][3] = {
        {{leaf(17), nullptr, nullptr}, {leaf(21), nullptr, nullptr}, {nullptr, nullptr, nullptr}},
        {{leaf(43), leaf(44), leaf(45)}, {leaf(49), leaf(50), leaf(51)}, {leaf(52), leaf(53), leaf(54)}},
        {{leaf(64), leaf(65), leaf(66)}, {leaf(70), leaf(71), leaf(72)}, {leaf(73), leaf(74), leaf(75)}}};
    const float* YW[3][3][3] = {
        {{leaf(18), leaf(19), leaf(20)}, {nullptr, nullptr, nullptr}, {leaf(22), leaf(23), leaf(24)}},
        {{leaf(46), leaf(47), leaf(48)}, {nullptr, nullptr, nullptr}, {leaf(55), leaf(56), leaf(57)}},
        {{leaf(67), leaf(68), leaf(69)}, {nullptr, nullptr, nullptr}, {leaf(76), leaf(77), leaf(78)}}};

    float* cur = nodeA;
    float* nxt = nodeB;
    long scatterThreads = (long)E * 32;
    int  scatterBlocks  = (int)((scatterThreads + 255) / 256);
    int  tiles   = (E + 15) / 16;
    int  gBlocks = (tiles + 1) / 2;      // 2 waves (tiles) per block

    for (int L = 0; L < 3; ++L) {
        int prevDim = D[L].prevDim;
        hipMemsetAsync(aggPass, 0, (size_t)N * prevDim * 4, stream);
        hipMemsetAsync(aggTP,   0, (size_t)N * 108 * 4, stream);
        scatter_pass_kernel<<<scatterBlocks, 256, 0, stream>>>(
            cur, senders, receivers, aggPass, E, prevDim);

        for (int t = 0; t < 3; ++t) {
            int c0, c1, c2, o0, o1, o2;
            if (L == 0) { c0 = 1; c1 = 0; c2 = 0; o0 = 0; o1 = 0; o2 = 0; }
            else if (t < 2) { c0 = 32; c1 = 8; c2 = 8; o0 = 0;  o1 = 88; o2 = 112; }
            else            { c0 = 32; c1 = 8; c2 = 8; o0 = 32; o1 = 64; o2 = 152; }

            xcoef_kernel<<<(E + 255) / 256, 256, 0, stream>>>(
                cur, senders, XW[L][t][0], XW[L][t][1], XW[L][t][2],
                c0, c1, c2, o0, o1, o2, prevDim, xcoefB, E);
            ycoef_kernel<<<(E + 255) / 256, 256, 0, stream>>>(
                shBuf, YW[L][t][0], YW[L][t][1], YW[L][t][2], ycoefB, E);
            gaunt_wmma_kernel<<<gBlocks, 64, 0, stream>>>(
                xcoefB, ycoefB, YGB, YGzB, gridW, zBuf, E);
            scatter_tp_kernel<<<scatterBlocks, 256, 0, stream>>>(
                zBuf, receivers, aggTP, E, t);
        }

        node_update_kernel<<<(N + 127) / 128, 128, 0, stream>>>(
            aggPass, aggTP, cur, nxt, D[L], Wl[L], N);
        float* tmp = cur; cur = nxt; nxt = tmp;
    }

    readout_kernel<<<1, 64, 0, stream>>>(cur, (float*)d_out, N, G);
}
